// NeuralHmm_62191126446241
// MI455X (gfx1250) — matrix operand order repaired and checked by reading, not compiled, not run on hardware
//
#include <hip/hip_runtime.h>
#include <hip/hip_bf16.h>
#include <math.h>

// ---------- constants ----------
#define NB   512   // batch
#define NT   20    // timesteps
#define NH   256   // hidden dim
#define NS   64    // s dim
#define NEMI 512   // emission hidden
#define NO   128   // output dim
#define NTD  512   // transition hidden (both layers)
#define NCIN 576   // NTD + NS

typedef __bf16 bf16_t;
typedef __attribute__((ext_vector_type(16))) __bf16 v16bf;
typedef __attribute__((ext_vector_type(8)))  float  v8f;
typedef __attribute__((ext_vector_type(8)))  unsigned int v8u;

// A-source modes for the fused GEMM
#define AMODE_BF16     0   // A is bf16, no normalization
#define AMODE_BN       1   // A is fp32, per-feature affine (scale,shift)
#define AMODE_BN_SPLIT 2   // fp32+affine for k<ksplit, raw fp32 from A1f for k>=ksplit

// ---------- RNG helpers (deterministic hash; no correctness oracle) ----------
__device__ __forceinline__ unsigned hash_u32(unsigned x) {
    x ^= x >> 16; x *= 0x7feb352du;
    x ^= x >> 15; x *= 0x846ca68bu;
    x ^= x >> 16; return x;
}
__device__ __forceinline__ float rng_uniform(unsigned h) {
    return ((h >> 8) + 0.5f) * (1.0f / 16777216.0f);
}
__device__ __forceinline__ float rng_normal(unsigned tag) {
    unsigned h1 = hash_u32(tag * 2654435761u + 0x9e3779b9u);
    unsigned h2 = hash_u32(h1 ^ 0x85ebca6bu);
    float u1 = rng_uniform(h1), u2 = rng_uniform(h2);
    return sqrtf(-2.0f * logf(u1)) * __cosf(6.2831853071795864f * u2);
}
__device__ __forceinline__ float rng_gumbel(unsigned tag) {
    float u = rng_uniform(hash_u32(tag * 0x9e3779b9u + 0x85ebca6bu));
    return -logf(-logf(u));
}
__device__ __forceinline__ unsigned pack2bf(float a, float b) {
    unsigned short ua = __builtin_bit_cast(unsigned short, (bf16_t)a);
    unsigned short ub = __builtin_bit_cast(unsigned short, (bf16_t)b);
    return ((unsigned)ub << 16) | ua;
}
// K index of fragment pair p for this half-wave (contiguous order k = 16*half + s, matching B)
__device__ __forceinline__ int frag_k(int p, int half) {
    return half * 16 + 2 * p;
}

// ---------- fused WMMA GEMM (templated A-mode; branch-free inner loops) ----------
// Y = act(normA(A) (MxK) * W(NxK)^T + bias)
// Epilogue: col < nsplit -> Y0/ldY0/act0 ; else -> Y1/ldY1/act1 (col-nsplit)
// act: 0 none, 1 relu, 2 elu(x)+1+1e-12
// Wave tile = 16x64 (4 accumulators, A fragment reused 4x). Block = 4 waves = 64x64.
template<int AMODE>
__global__ __launch_bounds__(128)
void gemm_wmma(const void* __restrict__ A0, int lda0,
               const float* __restrict__ bnSS,
               const float* __restrict__ A1f, int lda1, int ksplit,
               const bf16_t* __restrict__ W, const float* __restrict__ bias,
               float* __restrict__ Y0, int ldY0, int act0,
               float* __restrict__ Y1, int ldY1, int act1, int nsplit,
               int M, int N, int K)
{
    const int lane = threadIdx.x & 31;
    const int wave = threadIdx.x >> 5;
    const int nBlkN = N >> 6;
    const int blkM = blockIdx.x / nBlkN;
    const int blkN = blockIdx.x - blkM * nBlkN;
    const int m0 = blkM * 64 + wave * 16;
    const int n0 = blkN * 64;
    const int half = lane >> 4;
    const int l15  = lane & 15;
    const int row  = m0 + l15;

    v8f acc[4];
#pragma unroll
    for (int j = 0; j < 4; ++j) acc[j] = (v8f){0.f,0.f,0.f,0.f,0.f,0.f,0.f,0.f};

    const bf16_t* Ab = (const bf16_t*)A0 + (size_t)row * lda0;
    const float*  Af = (const float*) A0 + (size_t)row * lda0;
    const bf16_t* Bp[4];
#pragma unroll
    for (int j = 0; j < 4; ++j)
        Bp[j] = W + (size_t)(n0 + 16 * j + l15) * K + half * 16;

    const int kEnd1 = (AMODE == AMODE_BN_SPLIT) ? ksplit : K;

    // ---- main K loop (BN or bf16 region), branch-free ----
#pragma unroll 2
    for (int kk = 0; kk < kEnd1; kk += 32) {
        v8u au;
#pragma unroll
        for (int p = 0; p < 8; ++p) {
            int k = kk + frag_k(p, half);
            if (AMODE == AMODE_BF16) {
                au[p] = *reinterpret_cast<const unsigned*>(Ab + k);
            } else {
                float2 f = *reinterpret_cast<const float2*>(Af + k);
                float2 s0 = *reinterpret_cast<const float2*>(bnSS + 2 * k);
                float2 s1 = *reinterpret_cast<const float2*>(bnSS + 2 * k + 2);
                au[p] = pack2bf(fmaf(f.x, s0.x, s0.y), fmaf(f.y, s1.x, s1.y));
            }
        }
        v16bf a = __builtin_bit_cast(v16bf, au);
        v16bf b0 = *reinterpret_cast<const v16bf*>(Bp[0] + kk);
        v16bf b1 = *reinterpret_cast<const v16bf*>(Bp[1] + kk);
        v16bf b2 = *reinterpret_cast<const v16bf*>(Bp[2] + kk);
        v16bf b3 = *reinterpret_cast<const v16bf*>(Bp[3] + kk);
        if (kk + 32 < kEnd1) __builtin_prefetch(Bp[0] + kk + 32, 0, 0);

        acc[0] = __builtin_amdgcn_wmma_f32_16x16x32_bf16(false, a, false, b0, (short)0, acc[0], false, false);
        acc[1] = __builtin_amdgcn_wmma_f32_16x16x32_bf16(false, a, false, b1, (short)0, acc[1], false, false);
        acc[2] = __builtin_amdgcn_wmma_f32_16x16x32_bf16(false, a, false, b2, (short)0, acc[2], false, false);
        acc[3] = __builtin_amdgcn_wmma_f32_16x16x32_bf16(false, a, false, b3, (short)0, acc[3], false, false);
    }

    // ---- tail K loop: raw fp32 second region (`s` concat), branch-free ----
    if (AMODE == AMODE_BN_SPLIT) {
        const float* As = A1f + (size_t)row * lda1;
#pragma unroll 2
        for (int kk = ksplit; kk < K; kk += 32) {
            v8u au;
#pragma unroll
            for (int p = 0; p < 8; ++p) {
                int k = kk - ksplit + frag_k(p, half);
                float2 f = *reinterpret_cast<const float2*>(As + k);
                au[p] = pack2bf(f.x, f.y);
            }
            v16bf a = __builtin_bit_cast(v16bf, au);
            v16bf b0 = *reinterpret_cast<const v16bf*>(Bp[0] + kk);
            v16bf b1 = *reinterpret_cast<const v16bf*>(Bp[1] + kk);
            v16bf b2 = *reinterpret_cast<const v16bf*>(Bp[2] + kk);
            v16bf b3 = *reinterpret_cast<const v16bf*>(Bp[3] + kk);
            acc[0] = __builtin_amdgcn_wmma_f32_16x16x32_bf16(false, a, false, b0, (short)0, acc[0], false, false);
            acc[1] = __builtin_amdgcn_wmma_f32_16x16x32_bf16(false, a, false, b1, (short)0, acc[1], false, false);
            acc[2] = __builtin_amdgcn_wmma_f32_16x16x32_bf16(false, a, false, b2, (short)0, acc[2], false, false);
            acc[3] = __builtin_amdgcn_wmma_f32_16x16x32_bf16(false, a, false, b3, (short)0, acc[3], false, false);
        }
    }

    // ---- epilogue: D layout VGPR r -> row m0+half*8+r, col n0+16j+l15 ----
#pragma unroll
    for (int j = 0; j < 4; ++j) {
        int col = n0 + 16 * j + l15;
        float bcol = bias[col];
        int   act  = (col < nsplit) ? act0 : act1;
        float* Yp; size_t ld; int c;
        if (col < nsplit) { Yp = Y0; ld = ldY0; c = col; }
        else              { Yp = Y1; ld = ldY1; c = col - nsplit; }
#pragma unroll
        for (int r = 0; r < 8; ++r) {
            int orow = m0 + half * 8 + r;
            float v = acc[j][r] + bcol;
            if (act == 1)      v = fmaxf(v, 0.f);
            else if (act == 2) v = ((v > 0.f) ? v : (__expf(v) - 1.f)) + 1.0f + 1e-12f;
            Yp[(size_t)orow * ld + c] = v;
        }
    }
}

// ---------- BatchNorm stats -> per-feature affine (scale, shift) ----------
// scale = g / sqrt(var+eps); shift = be - mean*scale
__global__ __launch_bounds__(256)
void bn_stats(const float* __restrict__ X, int M, int ldX,
              const float* __restrict__ g, const float* __restrict__ be,
              float* __restrict__ ss)
{
    const int n = blockIdx.x;
    float s = 0.f, s2 = 0.f;
    for (int r = threadIdx.x; r < M; r += blockDim.x) {
        float v = X[(size_t)r * ldX + n];
        s += v; s2 += v * v;
    }
    __shared__ float sh[256], sh2[256];
    sh[threadIdx.x] = s; sh2[threadIdx.x] = s2;
    __syncthreads();
    for (int off = 128; off > 0; off >>= 1) {
        if (threadIdx.x < off) {
            sh[threadIdx.x]  += sh[threadIdx.x + off];
            sh2[threadIdx.x] += sh2[threadIdx.x + off];
        }
        __syncthreads();
    }
    if (threadIdx.x == 0) {
        float m  = sh[0] / (float)M;
        float v  = sh2[0] / (float)M - m * m;
        float sc = g[n] * rsqrtf(v + 1e-5f);
        ss[2 * n]     = sc;
        ss[2 * n + 1] = be[n] - m * sc;
    }
}

// ---------- fused: attention + Gumbel-max sample + var-BN + gather + reparam ----------
__global__ __launch_bounds__(256)
void sample_reparam(const float* __restrict__ hgru,
                    const float* __restrict__ sdpre, const float* __restrict__ ssv,
                    const float* __restrict__ out_mu, float* __restrict__ out_var,
                    bf16_t* __restrict__ htb, int t, unsigned seed)
{
    const int b    = blockIdx.x;
    const int wave = threadIdx.x >> 5;
    const int lane = threadIdx.x & 31;
    __shared__ float sc_sh[32];
    __shared__ int   pis_sh;

    const float* hb = hgru + (size_t)b * NT * NH;
    const float* hq = hb + (size_t)t * NH;
    for (int u = wave; u < t; u += 8) {
        const float* hu = hb + (size_t)u * NH;
        float p = 0.f;
        for (int k = lane; k < NH; k += 32) p += hq[k] * hu[k];
#pragma unroll
        for (int off = 16; off > 0; off >>= 1) p += __shfl_xor(p, off, 32);
        if (lane == 0)
            sc_sh[u] = p * 0.0625f + rng_gumbel(seed ^ (unsigned)(t * 40503 + b * 25 + u));
    }
    __syncthreads();
    if (threadIdx.x == 0) {
        float best = sc_sh[0]; int bu = 0;
        for (int u = 1; u < t; ++u) if (sc_sh[u] > best) { best = sc_sh[u]; bu = u; }
        pis_sh = bu;
    }
    __syncthreads();
    const int p = pis_sh;

    const int j = threadIdx.x;                  // NH == 256 == blockDim.x
    float bnv = fmaf(sdpre[(size_t)b * NH + j], ssv[2 * j], ssv[2 * j + 1]);
    out_var[(size_t)b * NT * NH + (size_t)t * NH + j] = bnv;
    float mu = out_mu[(size_t)b * NT * NH + (size_t)(p + 1) * NH + j];
    float va = (p == t - 1) ? bnv
                            : out_var[(size_t)b * NT * NH + (size_t)(p + 1) * NH + j];
    float n = rng_normal(seed ^ 0x51EDB01u ^ (unsigned)(t * 131071 + b * NH + j));
    htb[(size_t)b * NH + j] = (bf16_t)(mu + va * n);
}

// ---------- x_recon[:,t,:] += std * N(0,1) ----------
__global__ __launch_bounds__(256)
void add_noise(float* __restrict__ xr, const float* __restrict__ stdbuf,
               int t, unsigned seed)
{
    int idx = blockIdx.x * blockDim.x + threadIdx.x;
    if (idx >= NB * NO) return;
    int b = idx / NO, o = idx - b * NO;
    float* p = xr + (size_t)b * NT * NO + (size_t)t * NO + o;
    *p = *p + stdbuf[idx] * rng_normal(seed ^ (unsigned)(t * 524287 + idx));
}

// ---------- conversions / init ----------
__global__ __launch_bounds__(256)
void f32_to_bf16(const float* __restrict__ src, bf16_t* __restrict__ dst, int n)
{
    int i = blockIdx.x * blockDim.x + threadIdx.x;
    if (i < n) dst[i] = (bf16_t)src[i];
}

__global__ __launch_bounds__(256)
void zero_bf16(bf16_t* __restrict__ dst, int n)
{
    int i = blockIdx.x * blockDim.x + threadIdx.x;
    if (i < n) dst[i] = (bf16_t)0.0f;
}

__global__ __launch_bounds__(256)
void copy_f32(const float* __restrict__ src, float* __restrict__ dst, int n)
{
    int i = blockIdx.x * blockDim.x + threadIdx.x;
    if (i < n) dst[i] = src[i];
}

// t=0 init: xr[:,0,:] = x[:,0,:]; mu[:,0,:] = 0; var[:,0,:] = 0
__global__ __launch_bounds__(256)
void init_t0(const float* __restrict__ x, float* __restrict__ xr,
             float* __restrict__ mu, float* __restrict__ var)
{
    int idx = blockIdx.x * blockDim.x + threadIdx.x;
    if (idx < NB * NH) {
        int b = idx / NH, j = idx - b * NH;
        mu [(size_t)b * NT * NH + j] = 0.f;
        var[(size_t)b * NT * NH + j] = 0.f;
    }
    if (idx < NB * NO) {
        int b = idx / NO, o = idx - b * NO;
        xr[(size_t)b * NT * NO + o] = x[(size_t)b * NT * NO + o];
    }
}

// ---------- host launcher ----------
extern "C" void kernel_launch(void* const* d_in, const int* in_sizes, int n_in,
                              void* d_out, int out_size, void* d_ws, size_t ws_size,
                              hipStream_t stream)
{
    const float* x     = (const float*)d_in[0];
    const float* s     = (const float*)d_in[1];
    const float* h_gru = (const float*)d_in[2];
    const float* t_W1 = (const float*)d_in[6];
    const float* t_b1 = (const float*)d_in[7];
    const float* t_g1 = (const float*)d_in[8];
    const float* t_be1= (const float*)d_in[9];
    const float* t_W2 = (const float*)d_in[10];
    const float* t_b2 = (const float*)d_in[11];
    const float* t_g2 = (const float*)d_in[12];
    const float* t_be2= (const float*)d_in[13];
    const float* t_Wm = (const float*)d_in[14];
    const float* t_bm = (const float*)d_in[15];
    const float* t_Ws = (const float*)d_in[16];
    const float* t_bs = (const float*)d_in[17];
    const float* t_gs = (const float*)d_in[18];
    const float* t_bes= (const float*)d_in[19];
    const float* e_W1 = (const float*)d_in[20];
    const float* e_b1 = (const float*)d_in[21];
    const float* e_g1 = (const float*)d_in[22];
    const float* e_be1= (const float*)d_in[23];
    const float* e_Wm = (const float*)d_in[24];
    const float* e_bm = (const float*)d_in[25];
    const float* e_Ws = (const float*)d_in[26];
    const float* e_bs = (const float*)d_in[27];

    float* out_xr  = (float*)d_out;
    float* out_x   = out_xr + (size_t)NB * NT * NO;
    float* out_mu  = out_x  + (size_t)NB * NT * NO;
    float* out_var = out_mu + (size_t)NB * NT * NH;

    char* wp = (char*)d_ws;
    auto carve = [&](size_t bytes) -> char* {
        char* r = wp; wp += (bytes + 255) & ~(size_t)255; return r;
    };
    bf16_t* W1b    = (bf16_t*)carve((size_t)NTD  * NH   * 2);
    bf16_t* W2b    = (bf16_t*)carve((size_t)NTD  * NCIN * 2);
    bf16_t* WmWsb  = (bf16_t*)carve((size_t)(2*NH) * NTD * 2);   // [Wm; Ws] 512x512
    bf16_t* eW1b   = (bf16_t*)carve((size_t)NEMI * NH   * 2);
    bf16_t* eWmWsb = (bf16_t*)carve((size_t)(2*NO) * NEMI * 2);  // [eWm; eWs] 256x512
    float*  bmbs   = (float*) carve((size_t)(2*NH) * 4);         // [t_bm; t_bs]
    float*  ebmbs  = (float*) carve((size_t)(2*NO) * 4);         // [e_bm; e_bs]
    bf16_t* htb    = (bf16_t*)carve((size_t)NB * NH   * 2);      // current h row, bf16
    float*  h1pre  = (float*) carve((size_t)NB * NTD  * 4);
    float*  h2pre  = (float*) carve((size_t)NB * NTD  * 4);
    float*  sdpre  = (float*) carve((size_t)NB * NH   * 4);
    float*  eh1pre = (float*) carve((size_t)NB * NEMI * 4);
    float*  stdbuf = (float*) carve((size_t)NB * NO   * 4);
    float*  ss     = (float*) carve((size_t)NTD * 2 * 4);        // BN affine (scale,shift)
    float*  ssv    = (float*) carve((size_t)NH  * 2 * 4);        // sd BN affine (kept live)

    const unsigned SEED = 0xC0FFEE42u;
    auto cdiv = [](int a, int b) { return (a + b - 1) / b; };

    // ---- once-per-launch setup ----
    f32_to_bf16<<<cdiv(NTD*NH,   256), 256, 0, stream>>>(t_W1, W1b,  NTD*NH);
    f32_to_bf16<<<cdiv(NTD*NCIN, 256), 256, 0, stream>>>(t_W2, W2b,  NTD*NCIN);
    f32_to_bf16<<<cdiv(NH*NTD,   256), 256, 0, stream>>>(t_Wm, WmWsb,            NH*NTD);
    f32_to_bf16<<<cdiv(NH*NTD,   256), 256, 0, stream>>>(t_Ws, WmWsb + (size_t)NH*NTD, NH*NTD);
    f32_to_bf16<<<cdiv(NEMI*NH,  256), 256, 0, stream>>>(e_W1, eW1b, NEMI*NH);
    f32_to_bf16<<<cdiv(NO*NEMI,  256), 256, 0, stream>>>(e_Wm, eWmWsb,            NO*NEMI);
    f32_to_bf16<<<cdiv(NO*NEMI,  256), 256, 0, stream>>>(e_Ws, eWmWsb + (size_t)NO*NEMI, NO*NEMI);
    copy_f32<<<1, 256, 0, stream>>>(t_bm, bmbs,      NH);
    copy_f32<<<1, 256, 0, stream>>>(t_bs, bmbs + NH, NH);
    copy_f32<<<1, 128, 0, stream>>>(e_bm, ebmbs,      NO);
    copy_f32<<<1, 128, 0, stream>>>(e_bs, ebmbs + NO, NO);
    zero_bf16<<<cdiv(NB*NH, 256), 256, 0, stream>>>(htb, NB*NH);   // h_p_0 == 0
    copy_f32<<<cdiv(NB*NT*NO, 256), 256, 0, stream>>>(x, out_x, NB*NT*NO);
    init_t0<<<cdiv(NB*NH, 256), 256, 0, stream>>>(x, out_xr, out_mu, out_var);

    // ---- sequential timesteps (only the newest h row needs transition) ----
    for (int t = 1; t < NT; ++t) {
        // L1: relu(htb @ W1^T + b1) -> h1pre
        gemm_wmma<AMODE_BF16><<<(NB/64)*(NTD/64), 128, 0, stream>>>(
            htb, NH, nullptr, nullptr, 0, 0,
            W1b, t_b1, h1pre, NTD, 1, nullptr, 0, 0, NTD, NB, NTD, NH);
        bn_stats<<<NTD, 256, 0, stream>>>(h1pre, NB, NTD, t_g1, t_be1, ss);

        // L2: relu([BN(h1pre) | s] @ W2^T + b2) -> h2pre  (BN + concat fused into A-load)
        gemm_wmma<AMODE_BN_SPLIT><<<(NB/64)*(NTD/64), 128, 0, stream>>>(
            h1pre, NTD, ss, s, NS, NTD,
            W2b, t_b2, h2pre, NTD, 1, nullptr, 0, 0, NTD, NB, NTD, NCIN);
        bn_stats<<<NTD, 256, 0, stream>>>(h2pre, NB, NTD, t_g2, t_be2, ss);

        // mu & sd heads fused: BN(h2pre) @ [Wm;Ws]^T ; cols<256 -> mu_p_seq[:,t,:], cols>=256 -> elu+1 -> sdpre
        gemm_wmma<AMODE_BN><<<(NB/64)*((2*NH)/64), 128, 0, stream>>>(
            h2pre, NTD, ss, nullptr, 0, 0,
            WmWsb, bmbs, out_mu + (size_t)t * NH, NT*NH, 0, sdpre, NH, 2, NH,
            NB, 2*NH, NTD);
        bn_stats<<<NH, 256, 0, stream>>>(sdpre, NB, NH, t_gs, t_bes, ssv);

        // attention + Gumbel-max + var BN + gather + reparam -> htb, out_var[:,t,:]
        sample_reparam<<<NB, 256, 0, stream>>>(h_gru, sdpre, ssv, out_mu, out_var,
                                               htb, t, SEED);

        // emission L1: relu(htb @ eW1^T + e_b1) -> eh1pre
        gemm_wmma<AMODE_BF16><<<(NB/64)*(NEMI/64), 128, 0, stream>>>(
            htb, NH, nullptr, nullptr, 0, 0,
            eW1b, e_b1, eh1pre, NEMI, 1, nullptr, 0, 0, NEMI, NB, NEMI, NH);
        bn_stats<<<NEMI, 256, 0, stream>>>(eh1pre, NB, NEMI, e_g1, e_be1, ss);

        // emission heads fused: BN(eh1pre) @ [eWm;eWs]^T ; cols<128 -> x_recon[:,t,:], else elu+1 -> stdbuf
        gemm_wmma<AMODE_BN><<<(NB/64)*((2*NO)/64), 128, 0, stream>>>(
            eh1pre, NEMI, ss, nullptr, 0, 0,
            eWmWsb, ebmbs, out_xr + (size_t)t * NO, NT*NO, 0, stdbuf, NO, 2, NO,
            NB, 2*NO, NEMI);
        add_noise<<<cdiv(NB*NO, 256), 256, 0, stream>>>(out_xr, stdbuf, t, SEED ^ 0x7654321u);
    }
}
